// EdgeConvBlock_68375879352644
// MI455X (gfx1250) — matrix operand order repaired and checked by reading, not compiled, not run on hardware
//
#include <hip/hip_runtime.h>

#define IN_FEAT 128
#define TE 128          // edges per block (8 waves x 16-row M tiles)
#define XS 272          // X tile row stride in f16 (256 + 16 pad)
#define HS 136          // H tile row stride in f16 (128 + 8 pad)

typedef __attribute__((ext_vector_type(16))) _Float16 v16h;
typedef __attribute__((ext_vector_type(8)))  _Float16 v8h;
typedef __attribute__((ext_vector_type(8)))  float    v8f;

// A-fragment loader: a lane holds 16 contiguous f16 of its row at
// K+0..15 (lane>=16 offsets by +16), matching the B fill.
static __device__ __forceinline__ v16h ld_afrag(const _Float16* p) {
  v8h lo = *(const v8h*)(p);
  v8h hi = *(const v8h*)(p + 8);
  return __builtin_shufflevector(lo, hi, 0,1,2,3,4,5,6,7,8,9,10,11,12,13,14,15);
}

__global__ void cvt_f16_kernel(const float* __restrict__ src,
                               _Float16* __restrict__ dst, int n) {
  int i = blockIdx.x * 256 + threadIdx.x;
  if (i < n) dst[i] = (_Float16)src[i];
}

__global__ void copy_kernel(const float4* __restrict__ src,
                            float4* __restrict__ dst, int n4) {
  int i = blockIdx.x * 256 + threadIdx.x;
  if (i < n4) dst[i] = src[i];
}

__global__ void relu_kernel(float4* __restrict__ buf, int n4) {
  int i = blockIdx.x * 256 + threadIdx.x;
  if (i < n4) {
    float4 v = buf[i];
    v.x = fmaxf(v.x, 0.f);
    v.y = fmaxf(v.y, 0.f);
    v.z = fmaxf(v.z, 0.f);
    v.w = fmaxf(v.w, 0.f);
    buf[i] = v;
  }
}

__global__ __launch_bounds__(256) void edgeconv_kernel(
    const float* __restrict__ features,       // [N,128] f32
    const _Float16* __restrict__ W0h,         // [128,256] f16 row-major
    const float* __restrict__ b0,             // [128]
    const _Float16* __restrict__ W1h,         // [128,128] f16 row-major
    const float* __restrict__ b1,             // [128]
    const int* __restrict__ ind_p1,
    const int* __restrict__ ind_p2,
    float* __restrict__ out,                  // pre-loaded with features; scatter-add target
    int n_edges) {
  __shared__ _Float16 Xs[TE * XS];            // [128,256] concat tile (f16, padded)
  __shared__ _Float16 Hs[TE * HS];            // [128,128] hidden tile (f16, padded)

  const int tid = threadIdx.x;
  const int blk = blockIdx.x;

  // ---- Phase 1: gather + diff + concat into LDS (f32 -> f16) ----
  {
    const int s = tid >> 1;                   // edge slot 0..127
    const int h = tid & 1;                    // which 64-feature half
    const int eg = blk * TE + s;
    int p1 = 0, p2 = 0;
    if (eg < n_edges) { p1 = ind_p1[eg]; p2 = ind_p2[eg]; }
    const float4* f1 = (const float4*)(features + (size_t)p1 * IN_FEAT) + h * 16;
    const float4* f2 = (const float4*)(features + (size_t)p2 * IN_FEAT) + h * 16;
    _Float16* xr = &Xs[s * XS + h * 64];
#pragma unroll
    for (int j = 0; j < 16; ++j) {
      float4 a = f1[j];
      float4 b = f2[j];
      xr[4 * j + 0] = (_Float16)a.x;
      xr[4 * j + 1] = (_Float16)a.y;
      xr[4 * j + 2] = (_Float16)a.z;
      xr[4 * j + 3] = (_Float16)a.w;
      xr[128 + 4 * j + 0] = (_Float16)(b.x - a.x);
      xr[128 + 4 * j + 1] = (_Float16)(b.y - a.y);
      xr[128 + 4 * j + 2] = (_Float16)(b.z - a.z);
      xr[128 + 4 * j + 3] = (_Float16)(b.w - a.w);
    }
  }
  __syncthreads();

  const int w  = tid >> 5;                    // wave id -> M tile (16 edges)
  const int lm = tid & 15;                    // lane % 16  (N column / A row)
  const int lh = (tid >> 4) & 1;              // lane high half (K phase)

  // ---- Phase 2: GEMM1  H = relu(X @ W0^T + b0), X:[128,256] W0:[128,256] ----
  {
    const _Float16* arow = &Xs[(w * 16 + lm) * XS + lh * 16];
    v16h A[8];
#pragma unroll
    for (int kt = 0; kt < 8; ++kt) A[kt] = ld_afrag(arow + kt * 32);

#pragma unroll
    for (int nt = 0; nt < 8; ++nt) {
      const int n = nt * 16 + lm;
      const _Float16* bp = W0h + (size_t)n * 256 + lh * 16;  // B: lane=N, 16 contiguous K
      v8f acc = {};
#pragma unroll
      for (int kt = 0; kt < 8; ++kt) {
        v16h B = *(const v16h*)(bp + kt * 32);
        acc = __builtin_amdgcn_wmma_f32_16x16x32_f16(
            false, A[kt], false, B, (short)0, acc, false, false);
      }
      const float bias = b0[n];
#pragma unroll
      for (int r = 0; r < 8; ++r) {
        float v = acc[r] + bias;
        v = v > 0.f ? v : 0.f;
        Hs[(w * 16 + lh * 8 + r) * HS + n] = (_Float16)v;   // C/D layout: M=r(+8), N=lm
      }
    }
  }
  __syncthreads();

  // ---- Phase 3: GEMM2  Y = relu(H @ W1^T + b1), then scatter-add by ind_p1 ----
  {
    const _Float16* arow = &Hs[(w * 16 + lm) * HS + lh * 16];
    v16h A[4];
#pragma unroll
    for (int kt = 0; kt < 4; ++kt) A[kt] = ld_afrag(arow + kt * 32);

    int nodes[8];
#pragma unroll
    for (int r = 0; r < 8; ++r) {
      const int eg = blk * TE + w * 16 + lh * 8 + r;
      nodes[r] = (eg < n_edges) ? ind_p1[eg] : -1;
    }

#pragma unroll
    for (int nt = 0; nt < 8; ++nt) {
      const int n = nt * 16 + lm;
      const _Float16* bp = W1h + (size_t)n * 128 + lh * 16;
      v8f acc = {};
#pragma unroll
      for (int kt = 0; kt < 4; ++kt) {
        v16h B = *(const v16h*)(bp + kt * 32);
        acc = __builtin_amdgcn_wmma_f32_16x16x32_f16(
            false, A[kt], false, B, (short)0, acc, false, false);
      }
      const float bias = b1[n];
#pragma unroll
      for (int r = 0; r < 8; ++r) {
        if (nodes[r] >= 0) {
          float v = acc[r] + bias;
          v = v > 0.f ? v : 0.f;
          atomicAdd(out + (size_t)nodes[r] * IN_FEAT + n, v);
        }
      }
    }
  }
}

extern "C" void kernel_launch(void* const* d_in, const int* in_sizes, int n_in,
                              void* d_out, int out_size, void* d_ws, size_t ws_size,
                              hipStream_t stream) {
  const float* features = (const float*)d_in[0];
  const float* W0       = (const float*)d_in[1];
  const float* b0       = (const float*)d_in[2];
  const float* W1       = (const float*)d_in[3];
  const float* b1       = (const float*)d_in[4];
  const int*   p1       = (const int*)d_in[5];
  const int*   p2       = (const int*)d_in[6];
  float* out = (float*)d_out;

  const int nfeat = in_sizes[0];   // N * 128
  const int nW0   = in_sizes[1];   // 128*256
  const int nW1   = in_sizes[3];   // 128*128
  const int E     = in_sizes[5];

  _Float16* W0h = (_Float16*)d_ws;
  _Float16* W1h = (_Float16*)((char*)d_ws + (size_t)nW0 * sizeof(_Float16));

  cvt_f16_kernel<<<(nW0 + 255) / 256, 256, 0, stream>>>(W0, W0h, nW0);
  cvt_f16_kernel<<<(nW1 + 255) / 256, 256, 0, stream>>>(W1, W1h, nW1);

  const int n4 = nfeat / 4;
  copy_kernel<<<(n4 + 255) / 256, 256, 0, stream>>>((const float4*)features,
                                                    (float4*)out, n4);

  edgeconv_kernel<<<(E + TE - 1) / TE, 256, 0, stream>>>(
      features, W0h, b0, W1h, b1, p1, p2, out, E);

  relu_kernel<<<(n4 + 255) / 256, 256, 0, stream>>>((float4*)out, n4);
}
